// Encoder_12584254177689
// MI455X (gfx1250) — matrix operand order repaired and checked by reading, not compiled, not run on hardware
//
#include <hip/hip_runtime.h>
#include <hip/hip_bf16.h>
#include <stdint.h>

typedef __attribute__((ext_vector_type(16))) __bf16        v16bf;
typedef __attribute__((ext_vector_type(8)))  float         v8f;
typedef __attribute__((ext_vector_type(8)))  unsigned int  v8u;
typedef __attribute__((ext_vector_type(4)))  unsigned int  u32x4;
typedef __attribute__((ext_vector_type(8)))  unsigned int  u32x8;

__device__ __forceinline__ unsigned short f2bf(float f) {
  unsigned int u = __builtin_bit_cast(unsigned int, f);
  u += 0x7FFFu + ((u >> 16) & 1u);          // round-to-nearest-even
  return (unsigned short)(u >> 16);
}

__device__ __forceinline__ v8f vzero8() {
  v8f z = {0.f, 0.f, 0.f, 0.f, 0.f, 0.f, 0.f, 0.f};
  return z;
}

// Load a 16-element bf16 WMMA fragment from LDS as two aligned 16B chunks.
__device__ __forceinline__ v16bf load_frag_lds(const unsigned short* lds, int e0, int e1) {
  const unsigned int* w = (const unsigned int*)lds;
  uint4 a = *(const uint4*)(w + (e0 >> 1));
  uint4 b = *(const uint4*)(w + (e1 >> 1));
  v8u r = {a.x, a.y, a.z, a.w, b.x, b.y, b.z, b.w};
  return __builtin_bit_cast(v16bf, r);
}

// Same, but directly from global memory (two global_load_b128).
__device__ __forceinline__ v16bf load_frag_global(const unsigned short* p0,
                                                  const unsigned short* p1) {
  uint4 a = *(const uint4*)p0;
  uint4 b = *(const uint4*)p1;
  v8u r = {a.x, a.y, a.z, a.w, b.x, b.y, b.z, b.w};
  return __builtin_bit_cast(v16bf, r);
}

// ---------------------------------------------------------------------------
// Tensor Data Mover: async 2D tile (bf16) load Global -> LDS.
// Issued once by one wave; completion via TENSORcnt. D# per CDNA5 ISA:
//   group0: [1:0]=count=1, [63:32]=lds_addr, [120:64]=global_addr, [127:126]=2
//   group1: data_size=1(2B), tensor_dim0=tile_w, tensor_dim1=tile_h,
//           tile_dim0=tile_w, tile_dim1=tile_h, tensor_dim0_stride=row_stride,
//           optional LDS padding (pad_interval/pad_amount codes).
// ---------------------------------------------------------------------------
__device__ __forceinline__ void tdm_load_2d_bf16(unsigned lds_off, const void* gptr,
                                                 unsigned tile_w, unsigned tile_h,
                                                 unsigned row_stride_elems,
                                                 unsigned pad_enable,
                                                 unsigned pad_interval_code,
                                                 unsigned pad_amount_code) {
  unsigned long long ga = (unsigned long long)(size_t)gptr;
  u32x4 g0;
  g0[0] = 1u;                                            // count=1, user mode
  g0[1] = lds_off;                                       // lds_addr
  g0[2] = (unsigned)ga;                                  // global_addr[31:0]
  g0[3] = (unsigned)((ga >> 32) & 0x01FFFFFFull) | 0x80000000u;  // addr hi | type=2
  u32x8 g1;
  g1[0] = 0x00010000u | (pad_enable << 20) | (pad_interval_code << 22)
        | (pad_amount_code << 25);                       // data_size=2B
  g1[1] = (tile_w & 0xFFFFu) << 16;                      // tensor_dim0 lo16
  g1[2] = ((tile_w >> 16) & 0xFFFFu) | ((tile_h & 0xFFFFu) << 16); // dim0 hi, dim1 lo
  g1[3] = ((tile_h >> 16) & 0xFFFFu) | ((tile_w & 0xFFFFu) << 16); // dim1 hi, tile_dim0
  g1[4] = tile_h & 0xFFFFu;                              // tile_dim1 (tile_dim2=0)
  g1[5] = row_stride_elems;                              // tensor_dim0_stride lo32
  g1[6] = 0u;
  g1[7] = 0u;
  asm volatile("tensor_load_to_lds %0, %1" :: "s"(g0), "s"(g1) : "memory");
}

// ---------------------------------------------------------------------------
// Elementwise f32 -> bf16 cast (weights)
// ---------------------------------------------------------------------------
__global__ void cast_f32_bf16(const float* __restrict__ in,
                              unsigned short* __restrict__ out, int n) {
  int i = blockIdx.x * 256 + threadIdx.x;
  if (i < n) out[i] = f2bf(in[i]);
}

// ---------------------------------------------------------------------------
// Row LayerNorm, f32 in -> bf16 out. One block (256 thr) per row.
// ---------------------------------------------------------------------------
__global__ __launch_bounds__(256)
void layernorm_bf16(const float* __restrict__ x, const float* __restrict__ gamma,
                    const float* __restrict__ beta, unsigned short* __restrict__ out,
                    int D) {
  __shared__ float red[16];
  const int row = blockIdx.x;
  const float* xr = x + (size_t)row * D;
  float s = 0.f, s2 = 0.f;
  for (int i = threadIdx.x; i < D; i += 256) { float v = xr[i]; s += v; s2 += v * v; }
  #pragma unroll
  for (int m = 16; m >= 1; m >>= 1) { s += __shfl_xor(s, m, 32); s2 += __shfl_xor(s2, m, 32); }
  const int wave = threadIdx.x >> 5, lane = threadIdx.x & 31;
  if (lane == 0) { red[wave] = s; red[wave + 8] = s2; }
  __syncthreads();
  s = 0.f; s2 = 0.f;
  #pragma unroll
  for (int i = 0; i < 8; ++i) { s += red[i]; s2 += red[i + 8]; }
  const float mu   = s / (float)D;
  const float var  = s2 / (float)D - mu * mu;
  const float rstd = rsqrtf(var + 1e-5f);
  for (int i = threadIdx.x; i < D; i += 256) {
    float v = (xr[i] - mu) * rstd * gamma[i] + beta[i];
    out[(size_t)row * D + i] = f2bf(v);
  }
}

// ---------------------------------------------------------------------------
// WMMA bf16 GEMM: C[M,N] = A[M,K] * B[K,N] (+bias)(+GELU)(+residual)
// Block tile 128x128x32, 256 threads = 8 waves (2x4), each wave 64x32.
// A tile staged via Tensor Data Mover (async DMA, TENSORcnt);
// B tile transposed into N-major LDS by the waves (overlaps the DMA).
// ---------------------------------------------------------------------------
#define BM 128
#define BN 128
#define BK 32
#define BKP 40   // padded K-stride of N-major B tile (elements); 80B (16B aligned)

template <bool HAS_BIAS, bool DO_GELU, bool RESID, bool OUT_BF16>
__global__ __launch_bounds__(256)
void gemm_bf16(const unsigned short* __restrict__ A,
               const unsigned short* __restrict__ B,
               const float* __restrict__ bias,
               const float* __restrict__ resid,
               void* __restrict__ Cout,
               int Mdim, int Ndim, int Kdim) {
  (void)Mdim;
  __shared__ unsigned short ldsA[BM * BK];   // row-major [128][32], TDM target
  __shared__ unsigned short ldsB[BN * BKP];  // N-major   [128][40]

  const int t = threadIdx.x;
  const int lane = t & 31, wave = t >> 5;
  const int wm = wave >> 2, wn = wave & 3;
  const int halfSel = lane >> 4, l16 = lane & 15;
  const int m0 = blockIdx.y * BM, n0 = blockIdx.x * BN;
  const unsigned ldsA_off = (unsigned)(size_t)(&ldsA[0]);

  v8f acc[4][2];
  #pragma unroll
  for (int i = 0; i < 4; ++i)
    #pragma unroll
    for (int j = 0; j < 2; ++j) acc[i][j] = vzero8();

  const int bn_local = (t & 31) * 4, bk_base = t >> 5;

  for (int k0 = 0; k0 < Kdim; k0 += BK) {
    // ---- A tile: async TDM DMA (issued by wave 0 only; EXEC ignored) ----
    if (wave == 0) {
      tdm_load_2d_bf16(ldsA_off, A + (size_t)m0 * Kdim + k0,
                       BK, BM, (unsigned)Kdim, 0u, 0u, 0u);
    }
    // ---- B tile transposed into N-major layout (overlaps the DMA) ----
    #pragma unroll
    for (int i = 0; i < 4; ++i) {
      int kk = bk_base + i * 8;
      const unsigned short* src = B + (size_t)(k0 + kk) * Ndim + n0 + bn_local;
      uint2 v = *(const uint2*)src;
      ldsB[(bn_local + 0) * BKP + kk] = (unsigned short)(v.x & 0xffffu);
      ldsB[(bn_local + 1) * BKP + kk] = (unsigned short)(v.x >> 16);
      ldsB[(bn_local + 2) * BKP + kk] = (unsigned short)(v.y & 0xffffu);
      ldsB[(bn_local + 3) * BKP + kk] = (unsigned short)(v.y >> 16);
      __builtin_prefetch(src + (size_t)BK * Ndim, 0, 0);   // next K tile
    }
    if (wave == 0) __builtin_amdgcn_s_wait_tensorcnt(0);
    __syncthreads();

    v16bf bfr[2];
    #pragma unroll
    for (int ns = 0; ns < 2; ++ns) {
      int nB = wn * 32 + ns * 16 + l16;
      int e0 = nB * BKP + halfSel * 8;       // lanes<16: K=0..7 & 16..23
      bfr[ns] = load_frag_lds(ldsB, e0, e0 + 16);
    }
    #pragma unroll
    for (int ms = 0; ms < 4; ++ms) {
      int row = wm * 64 + ms * 16 + l16;
      int e0 = row * BK + halfSel * 8;       // lanes<16: K=0..7 & 16..23
      v16bf afr = load_frag_lds(ldsA, e0, e0 + 16);
      #pragma unroll
      for (int ns = 0; ns < 2; ++ns) {
        acc[ms][ns] = __builtin_amdgcn_wmma_f32_16x16x32_bf16(
            false, afr, false, bfr[ns], (short)0, acc[ms][ns], false, false);
      }
    }
    __syncthreads();
  }

  // ---- epilogue: C layout vgpr r -> row = r + 8*(lane>=16), col = lane%16 ----
  #pragma unroll
  for (int ms = 0; ms < 4; ++ms) {
    #pragma unroll
    for (int ns = 0; ns < 2; ++ns) {
      int col = n0 + wn * 32 + ns * 16 + l16;
      float bv = HAS_BIAS ? bias[col] : 0.0f;
      #pragma unroll
      for (int r = 0; r < 8; ++r) {
        int row = m0 + wm * 64 + ms * 16 + halfSel * 8 + r;
        float v = acc[ms][ns][r] + bv;
        if (DO_GELU) v = 0.5f * v * (1.0f + erff(v * 0.70710678118f));
        if (RESID)   v += resid[(size_t)row * Ndim + col];
        if (OUT_BF16) ((unsigned short*)Cout)[(size_t)row * Ndim + col] = f2bf(v);
        else          ((float*)Cout)[(size_t)row * Ndim + col] = v;
      }
    }
  }
}

// ---------------------------------------------------------------------------
// Flash attention with additive bias (softmax(qk/sqrt(d))*softmax(b) renorm
// == softmax(qk/sqrt(d)+b)). Block = 128 q rows (8 waves x 16), KV tile = 64.
// qkv layout: [B, N, 3, H, 64] bf16. Q frags load straight from global;
// K tile DMA'd by the TDM (with LDS pad to 144B rows); V transposed manually.
// ---------------------------------------------------------------------------
#define AN 2048
#define AH 12
#define AD 768
#define AQKV 2304   // 3*H*Dh, qkv row length in elements
#define QS 72       // padded LDS row stride (elements); 144B (16B aligned)

__global__ __launch_bounds__(256)
void attn_kernel(const unsigned short* __restrict__ qkv,
                 const float* __restrict__ bias,
                 unsigned short* __restrict__ outp) {
  __shared__ unsigned short ldsK[64 * QS];       // [kv_row][dim]  (B layout for Q@K^T)
  __shared__ unsigned short ldsVt[64 * QS];      // [dim][kv_row]  (B layout for P@V)
  __shared__ unsigned short ldsP[8 * 16 * QS];   // wave-private P staging

  const int t = threadIdx.x;
  const int lane = t & 31, wave = t >> 5;
  const int halfSel = lane >> 4, l16 = lane & 15;
  const int qt = blockIdx.x, h = blockIdx.y, b = blockIdx.z;
  const int q0 = qt * 128;
  const unsigned ldsK_off = (unsigned)(size_t)(&ldsK[0]);

  // ---- Q fragments straight from global (A-frag = two aligned 16B chunks) ----
  v16bf qf[2];
  {
    const unsigned short* qbase =
        qkv + (size_t)((b * AN + q0 + wave * 16 + l16)) * AQKV + h * 64;
    #pragma unroll
    for (int ks = 0; ks < 2; ++ks) {
      int e = ks * 32 + halfSel * 8;
      qf[ks] = load_frag_global(qbase + e, qbase + e + 16);
    }
  }

  v8f oacc[4];
  #pragma unroll
  for (int i = 0; i < 4; ++i) oacc[i] = vzero8();
  float mstate[8], lstate[8];
  #pragma unroll
  for (int r = 0; r < 8; ++r) { mstate[r] = -3.0e38f; lstate[r] = 0.f; }

  for (int kv0 = 0; kv0 < AN; kv0 += 64) {
    __syncthreads();   // previous iteration's K/V reads complete before overwrite
    // ---- K tile via TDM: 64x64 bf16, row stride 2304, LDS rows padded to 144B
    //      (pad_interval code 4 = 32 DWORDs, pad_amount code 3 = 4 DWORDs) ----
    if (wave == 0) {
      tdm_load_2d_bf16(ldsK_off,
                       qkv + (size_t)(b * AN + kv0) * AQKV + AH * 64 + h * 64,
                       64u, 64u, AQKV, 1u, 4u, 3u);
    }
    // ---- V tile transposed into [dim][kv] (overlaps the DMA) ----
    {
      int r = t >> 2, d0 = (t & 3) * 16;
      const unsigned short* vsrc =
          qkv + (size_t)(b * AN + kv0 + r) * AQKV + 2 * AH * 64 + h * 64 + d0;
      uint4 v0 = *(const uint4*)(vsrc);
      uint4 v1 = *(const uint4*)(vsrc + 8);
      unsigned vv[8] = {v0.x, v0.y, v0.z, v0.w, v1.x, v1.y, v1.z, v1.w};
      #pragma unroll
      for (int i = 0; i < 8; ++i) {
        ldsVt[(d0 + 2 * i + 0) * QS + r] = (unsigned short)(vv[i] & 0xffffu);
        ldsVt[(d0 + 2 * i + 1) * QS + r] = (unsigned short)(vv[i] >> 16);
      }
    }
    if (wave == 0) __builtin_amdgcn_s_wait_tensorcnt(0);
    __syncthreads();

    // ---- S = Q @ K^T (16 x 64 per wave) ----
    v8f sacc[4];
    #pragma unroll
    for (int i = 0; i < 4; ++i) sacc[i] = vzero8();
    #pragma unroll
    for (int ks = 0; ks < 2; ++ks) {
      #pragma unroll
      for (int nt = 0; nt < 4; ++nt) {
        int n = nt * 16 + l16;
        int e0 = n * QS + ks * 32 + halfSel * 8;
        v16bf kf = load_frag_lds(ldsK, e0, e0 + 16);
        sacc[nt] = __builtin_amdgcn_wmma_f32_16x16x32_bf16(
            false, qf[ks], false, kf, (short)0, sacc[nt], false, false);
      }
    }

    // ---- online softmax over 64 kv cols ----
    float sv[4][8], mloc[8];
    #pragma unroll
    for (int r = 0; r < 8; ++r) mloc[r] = -3.0e38f;
    #pragma unroll
    for (int nt = 0; nt < 4; ++nt) {
      int col = kv0 + nt * 16 + l16;
      #pragma unroll
      for (int r = 0; r < 8; ++r) {
        int row = q0 + wave * 16 + halfSel * 8 + r;
        float v = sacc[nt][r] * 0.125f + bias[((size_t)h * AN + row) * AN + col];
        sv[nt][r] = v;
        mloc[r] = fmaxf(mloc[r], v);
      }
    }
    #pragma unroll
    for (int r = 0; r < 8; ++r) {
      float m = mloc[r];
      m = fmaxf(m, __shfl_xor(m, 1, 32));
      m = fmaxf(m, __shfl_xor(m, 2, 32));
      m = fmaxf(m, __shfl_xor(m, 4, 32));
      m = fmaxf(m, __shfl_xor(m, 8, 32));
      float mn = fmaxf(mstate[r], m);
      float sc = __expf(mstate[r] - mn);
      mstate[r] = mn;
      float s = 0.f;
      #pragma unroll
      for (int nt = 0; nt < 4; ++nt) {
        float p = __expf(sv[nt][r] - mn);
        sv[nt][r] = p;
        s += p;
      }
      s += __shfl_xor(s, 1, 32);
      s += __shfl_xor(s, 2, 32);
      s += __shfl_xor(s, 4, 32);
      s += __shfl_xor(s, 8, 32);
      lstate[r] = lstate[r] * sc + s;
      #pragma unroll
      for (int nt = 0; nt < 4; ++nt) oacc[nt][r] *= sc;
    }

    // ---- restage P through wave-private LDS into A-fragment layout ----
    #pragma unroll
    for (int nt = 0; nt < 4; ++nt)
      #pragma unroll
      for (int r = 0; r < 8; ++r)
        ldsP[(wave * 16 + halfSel * 8 + r) * QS + nt * 16 + l16] = f2bf(sv[nt][r]);
    asm volatile("s_wait_dscnt 0" ::: "memory");

    // ---- O += P @ V ----
    #pragma unroll
    for (int ks = 0; ks < 2; ++ks) {
      int prow = wave * 16 + l16;
      int e0 = prow * QS + ks * 32 + halfSel * 8;
      v16bf pf = load_frag_lds(ldsP, e0, e0 + 16);
      #pragma unroll
      for (int nt = 0; nt < 4; ++nt) {
        int n = nt * 16 + l16;
        int ev = n * QS + ks * 32 + halfSel * 8;
        v16bf vf = load_frag_lds(ldsVt, ev, ev + 16);
        oacc[nt] = __builtin_amdgcn_wmma_f32_16x16x32_bf16(
            false, pf, false, vf, (short)0, oacc[nt], false, false);
      }
    }
  }

  // ---- normalize and emit bf16 [B*N, H*Dh] ----
  #pragma unroll
  for (int nt = 0; nt < 4; ++nt)
    #pragma unroll
    for (int r = 0; r < 8; ++r) {
      int row = q0 + wave * 16 + halfSel * 8 + r;
      int col = h * 64 + nt * 16 + l16;
      outp[((size_t)b * AN + row) * AD + col] = f2bf(oacc[nt][r] / lstate[r]);
    }
}

// ---------------------------------------------------------------------------
// Host launcher
// ---------------------------------------------------------------------------
extern "C" void kernel_launch(void* const* d_in, const int* in_sizes, int n_in,
                              void* d_out, int out_size, void* d_ws, size_t ws_size,
                              hipStream_t stream) {
  (void)in_sizes; (void)n_in; (void)out_size; (void)ws_size;
  const float* x     = (const float*)d_in[0];
  const float* Wqkv  = (const float*)d_in[1];
  const float* Wout  = (const float*)d_in[2];
  const float* g1    = (const float*)d_in[3];
  const float* be1   = (const float*)d_in[4];
  const float* g2    = (const float*)d_in[5];
  const float* be2   = (const float*)d_in[6];
  const float* W1    = (const float*)d_in[7];
  const float* b1    = (const float*)d_in[8];
  const float* W2    = (const float*)d_in[9];
  const float* b2    = (const float*)d_in[10];
  const float* abias = (const float*)d_in[11];
  float* outp = (float*)d_out;

  const int ROWS = 4096;   // B*N
  const int D = 768, QKVN = 2304, FF = 3072;

  char* ws = (char*)d_ws;
  size_t off = 0;
  auto alloc = [&](size_t bytes) {
    size_t o = off;
    off += (bytes + 255) & ~(size_t)255;
    return o;
  };
  unsigned short* hbf   = (unsigned short*)(ws + alloc((size_t)ROWS * D * 2));
  unsigned short* wqkvb = (unsigned short*)(ws + alloc((size_t)D * QKVN * 2));
  unsigned short* woutb = (unsigned short*)(ws + alloc((size_t)D * D * 2));
  unsigned short* w1b   = (unsigned short*)(ws + alloc((size_t)D * FF * 2));
  unsigned short* w2b   = (unsigned short*)(ws + alloc((size_t)FF * D * 2));
  unsigned short* qkv   = (unsigned short*)(ws + alloc((size_t)ROWS * QKVN * 2));
  unsigned short* attno = (unsigned short*)(ws + alloc((size_t)ROWS * D * 2));
  float*          x1    = (float*)(ws + alloc((size_t)ROWS * D * 4));
  unsigned short* h2    = (unsigned short*)(ws + alloc((size_t)ROWS * D * 2));
  unsigned short* ffh   = (unsigned short*)(ws + alloc((size_t)ROWS * FF * 2));

  // 1. pre-norm + cast weights to bf16
  layernorm_bf16<<<ROWS, 256, 0, stream>>>(x, g1, be1, hbf, D);
  cast_f32_bf16<<<(D * QKVN + 255) / 256, 256, 0, stream>>>(Wqkv, wqkvb, D * QKVN);
  cast_f32_bf16<<<(D * D + 255) / 256, 256, 0, stream>>>(Wout, woutb, D * D);
  cast_f32_bf16<<<(D * FF + 255) / 256, 256, 0, stream>>>(W1, w1b, D * FF);
  cast_f32_bf16<<<(FF * D + 255) / 256, 256, 0, stream>>>(W2, w2b, FF * D);

  // 2. qkv = LN(x) @ Wqkv  (bf16 out for the attention kernel)
  gemm_bf16<false, false, false, true>
      <<<dim3(QKVN / BN, ROWS / BM), 256, 0, stream>>>(hbf, wqkvb, nullptr, nullptr,
                                                       qkv, ROWS, QKVN, D);
  // 3. flash attention with bias
  attn_kernel<<<dim3(AN / 128, AH, 2), 256, 0, stream>>>(qkv, abias, attno);

  // 4. x1 = attn_out @ Wout + x
  gemm_bf16<false, false, true, false>
      <<<dim3(D / BN, ROWS / BM), 256, 0, stream>>>(attno, woutb, nullptr, x,
                                                    x1, ROWS, D, D);
  // 5. FFN pre-norm
  layernorm_bf16<<<ROWS, 256, 0, stream>>>(x1, g2, be2, h2, D);

  // 6. ffh = gelu(LN(x1) @ W1 + b1)  (bf16 out)
  gemm_bf16<true, true, false, true>
      <<<dim3(FF / BN, ROWS / BM), 256, 0, stream>>>(h2, w1b, b1, nullptr,
                                                     ffh, ROWS, FF, D);
  // 7. out = ffh @ W2 + b2 + x1
  gemm_bf16<true, false, true, false>
      <<<dim3(D / BN, ROWS / BM), 256, 0, stream>>>(ffh, w2b, b2, x1,
                                                    outp, ROWS, D, FF);
}
